// GraphNet_43164421325584
// MI455X (gfx1250) — matrix operand order repaired and checked by reading, not compiled, not run on hardware
//
#include <hip/hip_runtime.h>

typedef __attribute__((ext_vector_type(16))) __bf16 v16bf;
typedef __attribute__((ext_vector_type(8)))  __bf16 v8bf;
typedef __attribute__((ext_vector_type(8)))  float  v8f;

constexpr int B_   = 32;
constexpr int NPG  = 512;
constexpr int EPG  = 2048;
constexpr int N_   = B_ * NPG;   // 16384
constexpr int E_   = B_ * EPG;   // 65536
constexpr int ETOT = E_ + N_;    // 81920
constexpr int H1   = 512;        // preproc hidden / edge_total width
constexpr int H2   = 1024;       // GAT / conv width

#define DEVFN __device__ __forceinline__

DEVFN float leakyf(float v, float s) { return v >= 0.f ? v : s * v; }
DEVFN unsigned fkey(float f) {
    unsigned u = __float_as_uint(f);
    return (u & 0x80000000u) ? ~u : (u | 0x80000000u);
}
DEVFN float funkey(unsigned k) {
    return __uint_as_float((k & 0x80000000u) ? (k & 0x7fffffffu) : ~k);
}

union FragBF { v16bf v; v8bf h[2]; };

// ---------------------------------------------------------------------------
// Generic bf16 WMMA GEMM: out[M,Nout] = A[M,K] @ W[Nout,K]^T (+bias)(+act)
// Each wave computes a 32(M) x 64(N) tile via 8x v_wmma_f32_16x16x32_bf16
// per K-step.  MODE 0: store. MODE 1: fused GATv2 attention logit.
// MODE 2: fused GeneralConv edge-message scatter into conv[dst].
// Prefetches are unconditional: speculative global_prefetch_b8 is dropped
// silently on invalid addresses (ISA 7.3), so running one K-step past the
// end of a row is harmless and keeps the inner loop branch-free.
// ---------------------------------------------------------------------------
template <int MODE, bool HAS_BIAS, bool ACT>
__global__ __launch_bounds__(256) void gemm_bf16(
    const __bf16* __restrict__ A, const __bf16* __restrict__ W,
    const float* __restrict__ bias, float* __restrict__ out,
    int M, int K, int Nout,
    const int* __restrict__ srcF, const int* __restrict__ dstF,
    const float* __restrict__ X1, const float* __restrict__ X2,
    const float* __restrict__ attv, float* __restrict__ aux) {
    const int lane = threadIdx.x & 31;
    const int wave = threadIdx.x >> 5;
    const int nt64 = Nout >> 6;
    const int widx = blockIdx.x * 8 + wave;
    const int m0 = (widx / nt64) * 32;
    const int n0 = (widx % nt64) * 64;
    const int lh = lane >> 4;   // K-half selector
    const int lm = lane & 15;   // row (A) / col (B,D) within tile

    v8f acc[2][4];
#pragma unroll
    for (int mi = 0; mi < 2; ++mi)
#pragma unroll
        for (int s = 0; s < 4; ++s)
            acc[mi][s] = v8f{0.f, 0.f, 0.f, 0.f, 0.f, 0.f, 0.f, 0.f};

    const __bf16* arow0 = A + (size_t)(m0 + lm) * K + lh * 8;
    const __bf16* arow1 = A + (size_t)(m0 + 16 + lm) * K + lh * 8;
    const __bf16* wrow0 = W + (size_t)(n0 + lm) * K + lh * 8;

    for (int k0 = 0; k0 < K; k0 += 32) {
        FragBF a0, a1;
        a0.h[0] = *(const v8bf*)(arow0 + k0);       // K = k0 + lh*8 + 0..7
        a0.h[1] = *(const v8bf*)(arow0 + k0 + 16);  // K = k0 + 16 + lh*8 + 0..7
        a1.h[0] = *(const v8bf*)(arow1 + k0);
        a1.h[1] = *(const v8bf*)(arow1 + k0 + 16);
        __builtin_prefetch((const void*)(arow0 + k0 + 32), 0, 1);
        __builtin_prefetch((const void*)(arow1 + k0 + 32), 0, 1);
#pragma unroll
        for (int s = 0; s < 4; ++s) {
            const __bf16* wp = wrow0 + (size_t)s * 16 * K + k0;
            FragBF b;
            b.h[0] = *(const v8bf*)(wp);            // K = k0 + lh*8 + 0..7
            b.h[1] = *(const v8bf*)(wp + 16);       // K = k0 + 16 + lh*8 + 0..7
            __builtin_prefetch((const void*)(wp + 32), 0, 1);
            acc[0][s] = __builtin_amdgcn_wmma_f32_16x16x32_bf16(
                false, a0.v, false, b.v, (short)0, acc[0][s], false, false);
            acc[1][s] = __builtin_amdgcn_wmma_f32_16x16x32_bf16(
                false, a1.v, false, b.v, (short)0, acc[1][s], false, false);
        }
    }

#pragma unroll
    for (int mi = 0; mi < 2; ++mi) {
#pragma unroll
        for (int r = 0; r < 8; ++r) {
            const int m = m0 + mi * 16 + lh * 8 + r;
            if (MODE == 1) {  // attention logit: sum_n leaky(xl+xr+ew,0.2)*att
                const int sE = srcF[m], dE = dstF[m];
                float tsum = 0.f;
#pragma unroll
                for (int s = 0; s < 4; ++s) {
                    const int n = n0 + s * 16 + lm;
                    float t = X1[(size_t)sE * Nout + n] + X2[(size_t)dE * Nout + n] +
                              acc[mi][s][r];
                    tsum += leakyf(t, 0.2f) * attv[n];
                }
                tsum += __shfl_xor(tsum, 1, 32);
                tsum += __shfl_xor(tsum, 2, 32);
                tsum += __shfl_xor(tsum, 4, 32);
                tsum += __shfl_xor(tsum, 8, 32);
                if (lm == 0) atomicAdd(&aux[m], tsum);
            } else if (MODE == 2) {  // conv scatter: conv[dst] += acc+bce+hm[src]
                const int sE = srcF[m], dE = dstF[m];
#pragma unroll
                for (int s = 0; s < 4; ++s) {
                    const int n = n0 + s * 16 + lm;
                    float v = acc[mi][s][r] + bias[n] + X1[(size_t)sE * Nout + n];
                    atomicAdd(&out[(size_t)dE * Nout + n], v);
                }
            } else {
#pragma unroll
                for (int s = 0; s < 4; ++s) {
                    const int n = n0 + s * 16 + lm;
                    float v = acc[mi][s][r];
                    if (HAS_BIAS) v += bias[n];
                    if (ACT) v = leakyf(v, 0.01f);
                    out[(size_t)m * Nout + n] = v;
                }
            }
        }
    }
}

// ---------------------------------------------------------------------------
// Elementwise / scatter helper kernels
// ---------------------------------------------------------------------------
__global__ void k_fill0(float* p, size_t n) {
    size_t i = (size_t)blockIdx.x * 256 + threadIdx.x;
    if (i < n) p[i] = 0.f;
}

__global__ void k_cvt_bf16(const float* __restrict__ s, __bf16* __restrict__ d, size_t n) {
    size_t i = (size_t)blockIdx.x * 256 + threadIdx.x;
    if (i < n) d[i] = (__bf16)s[i];
}

// Wge [1024,511] -> bf16 [1024,512] zero-padded last column
__global__ void k_cvt_wge(const float* __restrict__ s, __bf16* __restrict__ d) {
    size_t i = (size_t)blockIdx.x * 256 + threadIdx.x;  // 1024*512
    int j = (int)(i & 511), o = (int)(i >> 9);
    d[i] = (j < 511) ? (__bf16)s[(size_t)o * 511 + j] : (__bf16)0.f;
}

__global__ void k_build_idx(const int* __restrict__ ei, int* __restrict__ srcF,
                            int* __restrict__ dstF) {
    int i = blockIdx.x * 256 + threadIdx.x;
    if (i >= ETOT) return;
    if (i < E_) { srcF[i] = ei[i]; dstF[i] = ei[E_ + i]; }
    else        { srcF[i] = i - E_; dstF[i] = i - E_; }
}

// h = leaky(x*Wn + bn) -> bf16 [N,512]
__global__ void k_node_mlp(const float* __restrict__ x, const float* __restrict__ Wn,
                           const float* __restrict__ bn, __bf16* __restrict__ h) {
    size_t i = (size_t)blockIdx.x * 256 + threadIdx.x;  // N*512
    int node = (int)(i >> 9), j = (int)(i & 511);
    h[i] = (__bf16)leakyf(x[node] * Wn[j] + bn[j], 0.01f);
}

// e = leaky(ea@We.T + be) -> bf16 [ETOT,512], col 511 = 0 (alpha slot)
__global__ void k_edge_mlp(const float* __restrict__ ea, const float* __restrict__ We,
                           const float* __restrict__ be, __bf16* __restrict__ et) {
    size_t i = (size_t)blockIdx.x * 256 + threadIdx.x;  // ETOT*512
    size_t r = i >> 9;
    int j = (int)(i & 511);
    if (j == 511) { et[i] = (__bf16)0.f; return; }
    float a0, a1, a2, a3, a4;
    if (r < (size_t)E_) {
        const float* p = ea + r * 5;
        a0 = p[0]; a1 = p[1]; a2 = p[2]; a3 = p[3]; a4 = p[4];
    } else { a0 = a1 = a2 = a3 = a4 = 1.f; }  // self-loop fill_value = 1
    const float* w = We + (size_t)j * 5;
    float v = a0 * w[0] + a1 * w[1] + a2 * w[2] + a3 * w[3] + a4 * w[4] + be[j];
    et[i] = (__bf16)leakyf(v, 0.01f);
}

// yg = leaky(y@Wg.T + bg) [32,512]
__global__ void k_global_mlp(const float* __restrict__ y, const float* __restrict__ Wg,
                             const float* __restrict__ bg, float* __restrict__ yg) {
    size_t i = (size_t)blockIdx.x * 256 + threadIdx.x;  // 32*512
    int b = (int)(i >> 9), j = (int)(i & 511);
    const float* yy = y + (size_t)b * 5;
    const float* w = Wg + (size_t)j * 5;
    float v = yy[0]*w[0] + yy[1]*w[1] + yy[2]*w[2] + yy[3]*w[3] + yy[4]*w[4] + bg[j];
    yg[i] = leakyf(v, 0.01f);
}

__global__ void k_segmax(const float* __restrict__ logit, const int* __restrict__ dstF,
                         unsigned* __restrict__ mxb) {
    int i = blockIdx.x * 256 + threadIdx.x;
    if (i < ETOT) atomicMax(&mxb[dstF[i]], fkey(logit[i]));
}

__global__ void k_expsum(const float* __restrict__ logit, const int* __restrict__ dstF,
                         const unsigned* __restrict__ mxb, float* __restrict__ ex,
                         float* __restrict__ denom, float* __restrict__ deg) {
    int i = blockIdx.x * 256 + threadIdx.x;
    if (i >= ETOT) return;
    int d = dstF[i];
    float e = __expf(logit[i] - funkey(mxb[d]));
    ex[i] = e;
    atomicAdd(&denom[d], e);
    atomicAdd(&deg[d], 1.f);
}

__global__ void k_alpha(const float* __restrict__ ex, const float* __restrict__ denom,
                        const int* __restrict__ dstF, float* __restrict__ alpha,
                        __bf16* __restrict__ et) {
    int i = blockIdx.x * 256 + threadIdx.x;
    if (i >= ETOT) return;
    float a = ex[i] / (denom[dstF[i]] + 1e-16f);
    alpha[i] = a;
    et[(size_t)i * H1 + 511] = (__bf16)a;  // edge_total = [e, alpha]
}

__global__ void k_gat_init(const float* __restrict__ bgat, float* __restrict__ gat) {
    size_t i = (size_t)blockIdx.x * 256 + threadIdx.x;  // N*1024
    gat[i] = bgat[i & 1023];
}

__global__ void k_gat_scatter(const float* __restrict__ alpha, const float* __restrict__ xl,
                              const int* __restrict__ srcF, const int* __restrict__ dstF,
                              float* __restrict__ gat) {
    size_t i = (size_t)blockIdx.x * 256 + threadIdx.x;  // ETOT*1024
    int r = (int)(i >> 10), j = (int)(i & 1023);
    atomicAdd(&gat[(size_t)dstF[r] * H2 + j], alpha[r] * xl[(size_t)srcF[r] * H2 + j]);
}

// conv = deg*hmi + gat (then edge messages are atomically scattered on top)
__global__ void k_conv_init(const float* __restrict__ deg, const float* __restrict__ hmi,
                            const float* __restrict__ gat, float* __restrict__ conv) {
    size_t i = (size_t)blockIdx.x * 256 + threadIdx.x;  // N*1024
    int n = (int)(i >> 10);
    conv[i] = deg[n] * hmi[i] + gat[i];
}

__global__ void k_pool_node(const float* __restrict__ conv, float* __restrict__ pn) {
    int b = blockIdx.x;                                // graph
    int j = blockIdx.y * 256 + threadIdx.x;            // col in [0,1024)
    int nz = blockIdx.z;                               // node chunk
    float s = 0.f;
    for (int n = 0; n < 64; ++n)
        s += conv[(size_t)(b * NPG + nz * 64 + n) * H2 + j];
    atomicAdd(&pn[b * H2 + j], s);
}

// module's contiguous 2560-row edge pooling (index-based, matches ebatch)
__global__ void k_pool_edge(const __bf16* __restrict__ et, float* __restrict__ pe) {
    int b = blockIdx.x;
    int j = blockIdx.y * 256 + threadIdx.x;            // col in [0,512)
    int rz = blockIdx.z;
    float s = 0.f;
    for (int r = 0; r < 320; ++r)
        s += (float)et[((size_t)b * 2560 + rz * 320 + r) * H1 + j];
    atomicAdd(&pe[b * H1 + j], s);
}

__global__ __launch_bounds__(256) void k_final(
    const float* __restrict__ pn, const float* __restrict__ pe,
    const float* __restrict__ yg, const float* __restrict__ W1,
    const float* __restrict__ b1, const float* __restrict__ W2,
    const float* __restrict__ b2, float* __restrict__ out) {
    int b = blockIdx.x, o = threadIdx.x;  // 256 hidden units
    const float* w = W1 + (size_t)o * 2048;
    float acc = b1[o];
    for (int k = 0; k < 1024; ++k) acc += (pn[b * H2 + k] * (1.f / 512.f)) * w[k];
    for (int k = 0; k < 512;  ++k) acc += (pe[b * H1 + k] * (1.f / 2560.f)) * w[1024 + k];
    for (int k = 0; k < 512;  ++k) acc += yg[b * H1 + k] * w[1536 + k];
    float t = leakyf(acc, 0.01f) * W2[o];
    __shared__ float red[256];
    red[o] = t;
    __syncthreads();
    for (int s = 128; s > 0; s >>= 1) {
        if (o < s) red[o] += red[o + s];
        __syncthreads();
    }
    if (o == 0) {
        float z = red[0] + b2[0];
        out[b] = 1.f / (1.f + __expf(-z));
    }
}

// ---------------------------------------------------------------------------
extern "C" void kernel_launch(void* const* d_in, const int* in_sizes, int n_in,
                              void* d_out, int out_size, void* d_ws, size_t ws_size,
                              hipStream_t stream) {
    (void)in_sizes; (void)n_in; (void)out_size; (void)ws_size;
    const float* x    = (const float*)d_in[0];
    const int*   ei   = (const int*)d_in[1];
    const float* eatt = (const float*)d_in[2];
    const float* y    = (const float*)d_in[3];
    const float* Wn = (const float*)d_in[5],  *bn = (const float*)d_in[6];
    const float* We = (const float*)d_in[7],  *be = (const float*)d_in[8];
    const float* Wg = (const float*)d_in[9],  *bg = (const float*)d_in[10];
    const float* Wl = (const float*)d_in[11], *bl = (const float*)d_in[12];
    const float* Wr = (const float*)d_in[13], *br = (const float*)d_in[14];
    const float* Wge = (const float*)d_in[15];
    const float* att = (const float*)d_in[16], *bgat = (const float*)d_in[17];
    const float* Wm = (const float*)d_in[18], *bm = (const float*)d_in[19];
    const float* Wmi = (const float*)d_in[20], *bmi = (const float*)d_in[21];
    const float* Wce = (const float*)d_in[22], *bce = (const float*)d_in[23];
    const float* W1 = (const float*)d_in[24], *b1 = (const float*)d_in[25];
    const float* W2 = (const float*)d_in[26], *b2 = (const float*)d_in[27];
    float* out = (float*)d_out;

    char* p = (char*)d_ws;
    auto alloc = [&](size_t bytes) -> char* {
        char* r = p;
        p += (bytes + 255) & ~(size_t)255;
        return r;
    };
    __bf16* h_bf   = (__bf16*)alloc((size_t)N_ * H1 * 2);
    __bf16* et_bf  = (__bf16*)alloc((size_t)ETOT * H1 * 2);
    __bf16* Wl_bf  = (__bf16*)alloc((size_t)H2 * H1 * 2);
    __bf16* Wr_bf  = (__bf16*)alloc((size_t)H2 * H1 * 2);
    __bf16* Wge_bf = (__bf16*)alloc((size_t)H2 * H1 * 2);
    __bf16* Wm_bf  = (__bf16*)alloc((size_t)H2 * H2 * 2);
    __bf16* Wmi_bf = (__bf16*)alloc((size_t)H2 * H2 * 2);
    __bf16* Wce_bf = (__bf16*)alloc((size_t)H2 * H1 * 2);
    float*  xl     = (float*)alloc((size_t)N_ * H2 * 4);
    float*  xr     = (float*)alloc((size_t)N_ * H2 * 4);
    float*  gat    = (float*)alloc((size_t)N_ * H2 * 4);
    __bf16* gat_bf = (__bf16*)alloc((size_t)N_ * H2 * 2);
    float*  hm     = (float*)alloc((size_t)N_ * H2 * 4);
    float*  hmi    = (float*)alloc((size_t)N_ * H2 * 4);
    float*  conv   = (float*)alloc((size_t)N_ * H2 * 4);
    float*  logit  = (float*)alloc((size_t)ETOT * 4);
    float*  ex     = (float*)alloc((size_t)ETOT * 4);
    float*  alpha  = (float*)alloc((size_t)ETOT * 4);
    unsigned* mxb  = (unsigned*)alloc((size_t)N_ * 4);
    float*  denom  = (float*)alloc((size_t)N_ * 4);
    float*  deg    = (float*)alloc((size_t)N_ * 4);
    int*    srcF   = (int*)alloc((size_t)ETOT * 4);
    int*    dstF   = (int*)alloc((size_t)ETOT * 4);
    float*  yg     = (float*)alloc((size_t)B_ * H1 * 4);
    float*  pn     = (float*)alloc((size_t)B_ * H2 * 4);
    float*  pe     = (float*)alloc((size_t)B_ * H1 * 4);

    auto blks = [](size_t n) { return (unsigned)((n + 255) / 256); };

    // zero-init accumulators (every call; buffers are not re-poisoned)
    k_fill0<<<blks(ETOT), 256, 0, stream>>>(logit, ETOT);
    k_fill0<<<blks(N_), 256, 0, stream>>>((float*)mxb, N_);  // key 0 == -inf
    k_fill0<<<blks(N_), 256, 0, stream>>>(denom, N_);
    k_fill0<<<blks(N_), 256, 0, stream>>>(deg, N_);
    k_fill0<<<blks((size_t)B_ * H2), 256, 0, stream>>>(pn, (size_t)B_ * H2);
    k_fill0<<<blks((size_t)B_ * H1), 256, 0, stream>>>(pe, (size_t)B_ * H1);

    k_build_idx<<<blks(ETOT), 256, 0, stream>>>(ei, srcF, dstF);

    // weight conversions to bf16
    k_cvt_bf16<<<blks((size_t)H2 * H1), 256, 0, stream>>>(Wl, Wl_bf, (size_t)H2 * H1);
    k_cvt_bf16<<<blks((size_t)H2 * H1), 256, 0, stream>>>(Wr, Wr_bf, (size_t)H2 * H1);
    k_cvt_wge<<<blks((size_t)H2 * H1), 256, 0, stream>>>(Wge, Wge_bf);
    k_cvt_bf16<<<blks((size_t)H2 * H2), 256, 0, stream>>>(Wm, Wm_bf, (size_t)H2 * H2);
    k_cvt_bf16<<<blks((size_t)H2 * H2), 256, 0, stream>>>(Wmi, Wmi_bf, (size_t)H2 * H2);
    k_cvt_bf16<<<blks((size_t)H2 * H1), 256, 0, stream>>>(Wce, Wce_bf, (size_t)H2 * H1);

    // preprocessing MLPs
    k_node_mlp<<<blks((size_t)N_ * H1), 256, 0, stream>>>(x, Wn, bn, h_bf);
    k_edge_mlp<<<blks((size_t)ETOT * H1), 256, 0, stream>>>(eatt, We, be, et_bf);
    k_global_mlp<<<blks((size_t)B_ * H1), 256, 0, stream>>>(y, Wg, bg, yg);

    auto gblk = [](int M, int Nout) { return (unsigned)((M / 32) * (Nout / 64) / 8); };

    // xl = h@Wl.T+bl ; xr = h@Wr.T+br
    gemm_bf16<0, true, false><<<gblk(N_, H2), 256, 0, stream>>>(
        h_bf, Wl_bf, bl, xl, N_, H1, H2, nullptr, nullptr, nullptr, nullptr, nullptr, nullptr);
    gemm_bf16<0, true, false><<<gblk(N_, H2), 256, 0, stream>>>(
        h_bf, Wr_bf, br, xr, N_, H1, H2, nullptr, nullptr, nullptr, nullptr, nullptr, nullptr);

    // fused:  logit[e] = sum_n leaky(xl[src]+xr[dst]+(e@Wge.T), 0.2) * att[n]
    gemm_bf16<1, false, false><<<gblk(ETOT, H2), 256, 0, stream>>>(
        et_bf, Wge_bf, nullptr, nullptr, ETOT, H1, H2, srcF, dstF, xl, xr, att, logit);

    // softmax over dst + degree
    k_segmax<<<blks(ETOT), 256, 0, stream>>>(logit, dstF, mxb);
    k_expsum<<<blks(ETOT), 256, 0, stream>>>(logit, dstF, mxb, ex, denom, deg);
    k_alpha<<<blks(ETOT), 256, 0, stream>>>(ex, denom, dstF, alpha, et_bf);

    // gat = segsum(alpha * xl[src]) + bgat
    k_gat_init<<<blks((size_t)N_ * H2), 256, 0, stream>>>(bgat, gat);
    k_gat_scatter<<<blks((size_t)ETOT * H2), 256, 0, stream>>>(alpha, xl, srcF, dstF, gat);
    k_cvt_bf16<<<blks((size_t)N_ * H2), 256, 0, stream>>>(gat, gat_bf, (size_t)N_ * H2);

    // hm / hmi
    gemm_bf16<0, true, false><<<gblk(N_, H2), 256, 0, stream>>>(
        gat_bf, Wm_bf, bm, hm, N_, H2, H2, nullptr, nullptr, nullptr, nullptr, nullptr, nullptr);
    gemm_bf16<0, true, false><<<gblk(N_, H2), 256, 0, stream>>>(
        gat_bf, Wmi_bf, bmi, hmi, N_, H2, H2, nullptr, nullptr, nullptr, nullptr, nullptr, nullptr);

    // conv = deg*hmi + gat, then scatter hm[src] + (edge_total@Wce.T + bce)
    k_conv_init<<<blks((size_t)N_ * H2), 256, 0, stream>>>(deg, hmi, gat, conv);
    gemm_bf16<2, true, false><<<gblk(ETOT, H2), 256, 0, stream>>>(
        et_bf, Wce_bf, bce, conv, ETOT, H1, H2, srcF, dstF, hm, nullptr, nullptr, nullptr);

    // pooling + head
    k_pool_node<<<dim3(B_, 4, 8), 256, 0, stream>>>(conv, pn);
    k_pool_edge<<<dim3(B_, 2, 8), 256, 0, stream>>>(et_bf, pe);
    k_final<<<B_, 256, 0, stream>>>(pn, pe, yg, W1, b1, W2, b2, out);
}
